// ARC30x30TrajectoryTransformer_64020782514544
// MI455X (gfx1250) — matrix operand order repaired and checked by reading, not compiled, not run on hardware
//
#include <hip/hip_runtime.h>
#include <stdint.h>

typedef __bf16 bf16;
typedef __attribute__((ext_vector_type(16))) __bf16 v16bf;
typedef __attribute__((ext_vector_type(8)))  __bf16 v8bf;
typedef __attribute__((ext_vector_type(8)))  float  v8f;

#define BB 4
#define TT 1806
#define EE 768
#define HH 12
#define HDQ 64
#define LL 6
#define VV 32
#define MM (BB*TT)            // 7224 rows
#define TP 1824               // T padded to multiple of 32
#define QTILES ((TT+15)/16)   // 113
#define STEPP 903
#define OBSN 900
#define NEGINF (-1e30f)
#define KC 128                // K-chunk staged per LDS buffer

static __device__ __forceinline__ v16bf pack16(v8bf lo, v8bf hi) {
  v16bf r;
#pragma unroll
  for (int i = 0; i < 8; ++i) { r[i] = lo[i]; r[8 + i] = hi[i]; }
  return r;
}

// -------------------- fp32 -> bf16 convert --------------------
__global__ __launch_bounds__(256) void cvt_kernel(const float* __restrict__ src,
                                                  bf16* __restrict__ dst, size_t n) {
  size_t i = (size_t)blockIdx.x * 256 + threadIdx.x;
  if (i < n) dst[i] = (bf16)src[i];
}

// -------------------- embedding --------------------
__global__ __launch_bounds__(256) void embed_kernel(
    const int* __restrict__ ids, const float* __restrict__ tok,
    const float* __restrict__ pos, const float* __restrict__ compe,
    const float* __restrict__ rowe, const float* __restrict__ cole,
    float* __restrict__ x) {
  int rt = blockIdx.x;            // 0..MM-1  (= b*TT + t)
  int t = rt % TT;
  int pis = t % STEPP;
  int comp = (pis < OBSN) ? 0 : (pis == OBSN ? 1 : (pis == OBSN + 1 ? 2 : 3));
  bool obs = pis < OBSN;
  int rowi = obs ? pis / 30 : 0;
  int coli = obs ? pis % 30 : 0;
  int id = ids[rt];
  id = min(max(id, 0), VV - 1);
#pragma unroll
  for (int i = 0; i < 3; ++i) {
    int e = threadIdx.x + 256 * i;
    float sp = 0.f;
    if (obs) sp = (e < 384) ? rowe[rowi * 384 + e] : cole[coli * 384 + (e - 384)];
    x[(size_t)rt * EE + e] = tok[(size_t)id * EE + e] + pos[(size_t)t * EE + e] +
                             compe[(size_t)comp * EE + e] + sp;
  }
}

// -------------------- layernorm (fp32 in, bf16 out) --------------------
__global__ __launch_bounds__(256) void ln_kernel(
    const float* __restrict__ x, const float* __restrict__ w,
    const float* __restrict__ b, bf16* __restrict__ out) {
  __shared__ float red[16];
  int row = blockIdx.x;
  int tid = threadIdx.x;
  float v[3];
#pragma unroll
  for (int i = 0; i < 3; ++i) v[i] = x[(size_t)row * EE + tid + 256 * i];
  float s = v[0] + v[1] + v[2];
  float sq = v[0] * v[0] + v[1] * v[1] + v[2] * v[2];
#pragma unroll
  for (int off = 1; off < 32; off <<= 1) { s += __shfl_xor(s, off); sq += __shfl_xor(sq, off); }
  int wv = tid >> 5, lane = tid & 31;
  if (lane == 0) { red[wv] = s; red[8 + wv] = sq; }
  __syncthreads();
  s = 0.f; sq = 0.f;
#pragma unroll
  for (int i = 0; i < 8; ++i) { s += red[i]; sq += red[8 + i]; }
  float mean = s * (1.0f / EE);
  float var = sq * (1.0f / EE) - mean * mean;
  float rstd = rsqrtf(var + 1e-5f);
#pragma unroll
  for (int i = 0; i < 3; ++i) {
    int e = tid + 256 * i;
    out[(size_t)row * EE + e] = (bf16)((v[i] - mean) * rstd * w[e] + b[e]);
  }
}

// -------------------- WMMA GEMM: D = A(MxK) * W(NxK)^T + bias --------------------
// A 16x128 stage is cooperatively DMA'd into LDS by all 8 waves (async-to-LDS),
// double buffered; each wave then runs 4 WMMAs per stage from LDS + global W.
// mode 0: store bf16; mode 1: gelu(exact) -> bf16; mode 2: resid(fp32) += D;
// mode 3: scatter bf16 into [b,h,d,s(padded TP)] layout (for V).
__global__ __launch_bounds__(256) void gemm_wmma_kernel(
    const bf16* __restrict__ A, const bf16* __restrict__ W,
    const float* __restrict__ bias, bf16* __restrict__ outBf,
    float* __restrict__ resid, int M, int N, int K, int mode) {
  __shared__ bf16 sA[2][16 * KC];
  int wave = threadIdx.x >> 5;
  int lane = threadIdx.x & 31;
  int mtile = blockIdx.x;
  int ntile = blockIdx.y * 8 + wave;
  int m0 = mtile * 16, n0 = ntile * 16;
  int kbA = (lane >> 4) << 3;   // unused; A frag K base is kbB (0 or 16)
  int kbB = (lane >> 4) << 4;   // B frag K base (0 or 16)

  // ---- cooperative async stage: 256 threads x 16B = 4KB = 16 rows x 128 cols ----
  int c = threadIdx.x;                       // chunk id 0..255
  int crow = c >> 4;                         // 16 chunks (of 8 bf16) per row
  int ckp = (c & 15) << 3;
  int arow_c = m0 + crow; if (arow_c >= M) arow_c = M - 1;
  const bf16* gsrc = A + (size_t)arow_c * K + ckp;

  auto issue = [&](int buf, int kc) {
    unsigned l = (unsigned)(uintptr_t)&sA[buf][c * 8];
    unsigned long long ga = (unsigned long long)(uintptr_t)(gsrc + kc);
    asm volatile("global_load_async_to_lds_b128 %0, %1, off" :: "v"(l), "v"(ga) : "memory");
  };

  issue(0, 0);
  asm volatile("s_wait_asynccnt 0" ::: "memory");
  __syncthreads();

  const bf16* Wp = W + (size_t)(n0 + (lane & 15)) * K + kbB;
  v8f acc = {};
  int buf = 0;
  for (int kc = 0; kc < K; kc += KC) {
    if (kc + KC < K) issue(buf ^ 1, kc + KC);
    __builtin_prefetch(Wp + kc + 2 * KC, 0, 1);
#pragma unroll
    for (int kk = 0; kk < KC; kk += 32) {
      const bf16* ap = &sA[buf][(lane & 15) * KC + kk + kbB];
      v16bf a = pack16(*(const v8bf*)ap, *(const v8bf*)(ap + 8));
      v16bf bm = *(const v16bf*)(Wp + kc + kk);
      acc = __builtin_amdgcn_wmma_f32_16x16x32_bf16(false, a, false, bm, (short)0, acc,
                                                    false, false);
    }
    asm volatile("s_wait_asynccnt 0" ::: "memory");
    __syncthreads();
    buf ^= 1;
  }

  int n = n0 + (lane & 15);
  float bvv = bias[n];
#pragma unroll
  for (int r = 0; r < 8; ++r) {
    int m = m0 + r + ((lane >> 4) << 3);
    if (m >= M) continue;
    float vI = acc[r] + bvv;
    if (mode == 1) vI = vI * 0.5f * (1.0f + erff(vI * 0.7071067811865476f));
    if (mode == 2) {
      resid[(size_t)m * N + n] += vI;
    } else if (mode == 3) {
      int bidx = m / TT, s = m - bidx * TT;
      int hh = n >> 6, d = n & 63;
      outBf[((size_t)((bidx * HH + hh) * HDQ + d)) * TP + s] = (bf16)vI;
    } else {
      outBf[(size_t)m * N + n] = (bf16)vI;
    }
  }
}

// -------------------- flash attention, one wave per 16-query tile --------------------
__global__ __launch_bounds__(32) void attn_kernel(
    const bf16* __restrict__ Q, const bf16* __restrict__ Kb,
    const bf16* __restrict__ VT, const int* __restrict__ amask,
    bf16* __restrict__ Y) {
  __shared__ float sP[16 * 32];
  const float scale = 0.125f;   // 1/sqrt(64)
  int lane = threadIdx.x;
  int bid = blockIdx.x;
  int qt = bid % QTILES;
  int bh = bid / QTILES;
  int h = bh % HH;
  int b = bh / HH;
  int q0 = qt * 16;
  int nidx = lane & 15;
  int kbA = (lane >> 4) << 3;
  int kbB = (lane >> 4) << 4;

  int qm = q0 + nidx; if (qm >= TT) qm = TT - 1;
  const bf16* qp = Q + ((size_t)(b * TT + qm)) * EE + h * HDQ;
  v16bf aq0 = pack16(*(const v8bf*)(qp + kbB),      *(const v8bf*)(qp + kbB + 8));
  v16bf aq1 = pack16(*(const v8bf*)(qp + 32 + kbB), *(const v8bf*)(qp + 40 + kbB));

  float mrow[8], lrow[8];
  v8f oacc[4] = {};
#pragma unroll
  for (int r = 0; r < 8; ++r) { mrow[r] = NEGINF; lrow[r] = 0.f; }

  int kmax = q0 + 16; if (kmax > TT) kmax = TT;
  for (int s0 = 0; s0 < kmax; s0 += 32) {
    int key0 = s0 + nidx, key1 = s0 + 16 + nidx;
    int kr0 = key0 < TT ? key0 : TT - 1;
    int kr1 = key1 < TT ? key1 : TT - 1;
    const bf16* kp0 = Kb + ((size_t)(b * TT + kr0)) * EE + h * HDQ;
    const bf16* kp1 = Kb + ((size_t)(b * TT + kr1)) * EE + h * HDQ;
    v16bf b00 = *(const v16bf*)(kp0 + kbB);
    v16bf b01 = *(const v16bf*)(kp0 + 32 + kbB);
    v16bf b10 = *(const v16bf*)(kp1 + kbB);
    v16bf b11 = *(const v16bf*)(kp1 + 32 + kbB);
    v8f sacc0 = {}, sacc1 = {};
    sacc0 = __builtin_amdgcn_wmma_f32_16x16x32_bf16(false, aq0, false, b00, (short)0, sacc0, false, false);
    sacc0 = __builtin_amdgcn_wmma_f32_16x16x32_bf16(false, aq1, false, b01, (short)0, sacc0, false, false);
    sacc1 = __builtin_amdgcn_wmma_f32_16x16x32_bf16(false, aq0, false, b10, (short)0, sacc1, false, false);
    sacc1 = __builtin_amdgcn_wmma_f32_16x16x32_bf16(false, aq1, false, b11, (short)0, sacc1, false, false);
    int am0 = (key0 < TT) ? amask[b * TT + key0] : 0;
    int am1 = (key1 < TT) ? amask[b * TT + key1] : 0;
#pragma unroll
    for (int r = 0; r < 8; ++r) {
      int m = q0 + r + kbA;
      float a0 = sacc0[r] * scale;
      float a1 = sacc1[r] * scale;
      if (key0 > m || !am0) a0 = NEGINF;
      if (key1 > m || !am1) a1 = NEGINF;
      float vmax = fmaxf(a0, a1);
#pragma unroll
      for (int off = 1; off < 16; off <<= 1) vmax = fmaxf(vmax, __shfl_xor(vmax, off));
      float mnew = fmaxf(mrow[r], vmax);
      float alpha = 1.f, p0 = 0.f, p1 = 0.f;
      if (mnew > NEGINF * 0.5f) {
        alpha = expf(mrow[r] - mnew);
        p0 = expf(a0 - mnew);
        p1 = expf(a1 - mnew);
        mrow[r] = mnew;
      }
      float rsum = p0 + p1;
#pragma unroll
      for (int off = 1; off < 16; off <<= 1) rsum += __shfl_xor(rsum, off);
      lrow[r] = lrow[r] * alpha + rsum;
#pragma unroll
      for (int nn = 0; nn < 4; ++nn) oacc[nn][r] *= alpha;
      sP[(r + kbA) * 32 + nidx] = p0;
      sP[(r + kbA) * 32 + 16 + nidx] = p1;
    }
    asm volatile("s_wait_dscnt 0" ::: "memory");
    v16bf ap;
    {
      const float* prow = &sP[(lane & 15) * 32];
#pragma unroll
      for (int i = 0; i < 8; ++i) {
        ap[i]     = (bf16)prow[kbB + i];
        ap[8 + i] = (bf16)prow[kbB + 8 + i];
      }
    }
    const bf16* vbase = VT + ((size_t)((b * HH + h) * HDQ)) * TP + s0 + kbB;
#pragma unroll
    for (int nn = 0; nn < 4; ++nn) {
      v16bf bvf = *(const v16bf*)(vbase + (size_t)(nn * 16 + nidx) * TP);
      oacc[nn] = __builtin_amdgcn_wmma_f32_16x16x32_bf16(false, ap, false, bvf, (short)0,
                                                         oacc[nn], false, false);
    }
  }
#pragma unroll
  for (int r = 0; r < 8; ++r) {
    int m = q0 + r + kbA;
    if (m >= TT) continue;
    float inv = 1.0f / lrow[r];
#pragma unroll
    for (int nn = 0; nn < 4; ++nn) {
      Y[((size_t)(b * TT + m)) * EE + h * HDQ + nn * 16 + nidx] = (bf16)(oacc[nn][r] * inv);
    }
  }
}

// -------------------- component-masked output heads --------------------
__global__ __launch_bounds__(256) void heads_kernel(
    const bf16* __restrict__ hf, const float* __restrict__ Wobs,
    const float* __restrict__ Wact, const float* __restrict__ Wrew,
    const float* __restrict__ Wval, float* __restrict__ out) {
  int idx = blockIdx.x * 256 + threadIdx.x;
  if (idx >= MM * VV) return;
  int col = idx & 31;
  int row = idx >> 5;
  int t = row % TT;
  int pis = t % STEPP;
  int comp = (pis < OBSN) ? 0 : (pis == OBSN ? 1 : (pis == OBSN + 1 ? 2 : 3));
  const float* w = nullptr;
  if (col < 11)      { if (comp == 0) w = Wobs + (size_t)col * EE; }
  else if (col < 16) { if (comp == 1) w = Wact + (size_t)(col - 11) * EE; }
  else if (col < 21) { if (comp == 2) w = Wrew + (size_t)(col - 16) * EE; }
  else if (col < 26) { if (comp == 3) w = Wval + (size_t)(col - 21) * EE; }
  float acc = 0.f;
  if (w) {
    const bf16* xr = hf + (size_t)row * EE;
    for (int k = 0; k < EE; ++k) acc += (float)xr[k] * w[k];
  }
  out[idx] = acc;
}

// -------------------- host orchestration --------------------
extern "C" void kernel_launch(void* const* d_in, const int* in_sizes, int n_in,
                              void* d_out, int out_size, void* d_ws, size_t ws_size,
                              hipStream_t stream) {
  (void)in_sizes; (void)n_in; (void)out_size; (void)ws_size;
  const int*   ids   = (const int*)d_in[0];
  const int*   amask = (const int*)d_in[1];
  const float* tok   = (const float*)d_in[2];
  const float* pos   = (const float*)d_in[3];
  const float* compe = (const float*)d_in[4];
  const float* rowe  = (const float*)d_in[5];
  const float* cole  = (const float*)d_in[6];
  const float* Wq = (const float*)d_in[7];  const float* bq = (const float*)d_in[8];
  const float* Wk = (const float*)d_in[9];  const float* bk = (const float*)d_in[10];
  const float* Wv = (const float*)d_in[11]; const float* bv = (const float*)d_in[12];
  const float* Wp = (const float*)d_in[13]; const float* bp = (const float*)d_in[14];
  const float* ln1w = (const float*)d_in[15]; const float* ln1b = (const float*)d_in[16];
  const float* ln2w = (const float*)d_in[17]; const float* ln2b = (const float*)d_in[18];
  const float* W1 = (const float*)d_in[19]; const float* b1 = (const float*)d_in[20];
  const float* W2 = (const float*)d_in[21]; const float* b2 = (const float*)d_in[22];
  const float* lnfw = (const float*)d_in[23]; const float* lnfb = (const float*)d_in[24];
  const float* Wobs = (const float*)d_in[25]; const float* Wact = (const float*)d_in[26];
  const float* Wrew = (const float*)d_in[27]; const float* Wval = (const float*)d_in[28];
  float* out = (float*)d_out;

  uintptr_t cur = (uintptr_t)d_ws;
  auto alloc = [&](size_t bytes) -> void* {
    uintptr_t r = (cur + 255) & ~(uintptr_t)255;
    cur = r + bytes;
    return (void*)r;
  };
  const size_t nEE = (size_t)LL * EE * EE;
  const size_t n4E = (size_t)LL * 4 * EE * EE;
  bf16* wqb = (bf16*)alloc(nEE * 2);
  bf16* wkb = (bf16*)alloc(nEE * 2);
  bf16* wvb = (bf16*)alloc(nEE * 2);
  bf16* wpb = (bf16*)alloc(nEE * 2);
  bf16* w1b = (bf16*)alloc(n4E * 2);
  bf16* w2b = (bf16*)alloc(n4E * 2);
  float* x  = (float*)alloc((size_t)MM * EE * 4);
  bf16* hb  = (bf16*)alloc((size_t)MM * EE * 2);
  bf16* qb  = (bf16*)alloc((size_t)MM * EE * 2);
  bf16* kbf = (bf16*)alloc((size_t)MM * EE * 2);
  bf16* vt  = (bf16*)alloc((size_t)BB * HH * HDQ * TP * 2);
  bf16* yb  = (bf16*)alloc((size_t)MM * EE * 2);
  bf16* h2  = (bf16*)alloc((size_t)MM * 4 * EE * 2);

  auto cvt = [&](const float* s, bf16* dptr, size_t n) {
    cvt_kernel<<<(unsigned)((n + 255) / 256), 256, 0, stream>>>(s, dptr, n);
  };
  cvt(Wq, wqb, nEE); cvt(Wk, wkb, nEE); cvt(Wv, wvb, nEE); cvt(Wp, wpb, nEE);
  cvt(W1, w1b, n4E); cvt(W2, w2b, n4E);

  embed_kernel<<<MM, 256, 0, stream>>>(ids, tok, pos, compe, rowe, cole, x);

  const int MT = (MM + 15) / 16;           // 452
  dim3 gE(MT, EE / 128);                   // N=768 -> 48 ntiles, grid.y=6
  dim3 g4(MT, (4 * EE) / 128);             // N=3072 -> grid.y=24

  for (int l = 0; l < LL; ++l) {
    ln_kernel<<<MM, 256, 0, stream>>>(x, ln1w + (size_t)l * EE, ln1b + (size_t)l * EE, hb);
    gemm_wmma_kernel<<<gE, 256, 0, stream>>>(hb, wqb + (size_t)l * EE * EE,
                                             bq + (size_t)l * EE, qb, nullptr, MM, EE, EE, 0);
    gemm_wmma_kernel<<<gE, 256, 0, stream>>>(hb, wkb + (size_t)l * EE * EE,
                                             bk + (size_t)l * EE, kbf, nullptr, MM, EE, EE, 0);
    gemm_wmma_kernel<<<gE, 256, 0, stream>>>(hb, wvb + (size_t)l * EE * EE,
                                             bv + (size_t)l * EE, vt, nullptr, MM, EE, EE, 3);
    attn_kernel<<<BB * HH * QTILES, 32, 0, stream>>>(qb, kbf, vt, amask, yb);
    gemm_wmma_kernel<<<gE, 256, 0, stream>>>(yb, wpb + (size_t)l * EE * EE,
                                             bp + (size_t)l * EE, nullptr, x, MM, EE, EE, 2);
    ln_kernel<<<MM, 256, 0, stream>>>(x, ln2w + (size_t)l * EE, ln2b + (size_t)l * EE, hb);
    gemm_wmma_kernel<<<g4, 256, 0, stream>>>(hb, w1b + (size_t)l * 4 * EE * EE,
                                             b1 + (size_t)l * 4 * EE, h2, nullptr, MM, 4 * EE, EE, 1);
    gemm_wmma_kernel<<<gE, 256, 0, stream>>>(h2, w2b + (size_t)l * EE * 4 * EE,
                                             b2 + (size_t)l * EE, nullptr, x, MM, EE, 4 * EE, 2);
  }
  ln_kernel<<<MM, 256, 0, stream>>>(x, lnfw, lnfb, hb);
  heads_kernel<<<(MM * VV + 255) / 256, 256, 0, stream>>>(hb, Wobs, Wact, Wrew, Wval, out);
}
